// TopKSAE_6597069766699
// MI455X (gfx1250) — matrix operand order repaired and checked by reading, not compiled, not run on hardware
//
#include <hip/hip_runtime.h>
#include <hip/hip_bf16.h>

// ---------------- problem constants (match reference) ----------------
#define ACT_DIM   4096
#define DICT_SIZE 32768
#define TOPK      64
#define NTOK      4096

// GEMM blocking: block = 8 waves; block tile = 16 tokens x 128 cols.
#define BROWS      128            // B rows (output cols) per block
#define KC         32             // K chunk per WMMA step
#define BPAD       4              // pad DWORDs per LDS row (bank skew)
#define BSTRIDE    (KC + BPAD)    // 36 dwords = 144 B row stride in LDS

typedef __attribute__((ext_vector_type(16))) __bf16 v16bf;
typedef __attribute__((ext_vector_type(8)))  float  v8f;
typedef unsigned int v4u __attribute__((ext_vector_type(4)));
typedef unsigned int v8u __attribute__((ext_vector_type(8)));

// Split an fp32 value into hi/lo bf16 halves (Markidis split).
__device__ __forceinline__ void split_bf16(float x, __bf16 &h, __bf16 &l) {
    __bf16 hh = (__bf16)x;
    h = hh;
    l = (__bf16)(x - (float)hh);
}

// ---------------------------------------------------------------------
// TDM: DMA a [BROWS x KC] fp32 tile (row stride = ld elements) from global
// memory into LDS at lds_byte_off, with 4-DWORD padding every 32 DWORDs
// so the LDS layout is [BROWS][BSTRIDE] floats. 2D descriptor (groups 0+1).
// ---------------------------------------------------------------------
__device__ __forceinline__ void tdm_load_tile(const float* gsrc,
                                              unsigned lds_byte_off,
                                              unsigned ld /* tensor_dim0 & stride */)
{
    unsigned long long ga = (unsigned long long)(const void*)gsrc;

    v4u g0;
    g0[0] = 1u;                                         // count=1, user mode
    g0[1] = lds_byte_off;                               // lds_addr
    g0[2] = (unsigned)(ga & 0xFFFFFFFFu);               // global_addr[31:0]
    g0[3] = (unsigned)((ga >> 32) & 0x01FFFFFFu)        // global_addr[56:32]
          | (2u << 30);                                 // type = 2 (image)

    v8u g1;
    g1[0] = (2u << 16)                                  // data_size = 4 B
          | (1u << 20)                                  // pad_enable
          | (4u << 22)                                  // pad_interval: 32 DW
          | (3u << 25);                                 // pad_amount:   4 DW
    g1[1] = (ld & 0xFFFFu) << 16;                       // tensor_dim0[15:0]
    g1[2] = (ld >> 16) | (((unsigned)BROWS & 0xFFFFu) << 16); // dim0 hi | dim1 lo
    g1[3] = ((unsigned)KC << 16);                       // dim1 hi=0 | tile_dim0=32
    g1[4] = (unsigned)BROWS;                            // tile_dim1=128, tile_dim2=0
    g1[5] = ld;                                         // tensor_dim0_stride[31:0]
    g1[6] = 0u;                                         // stride hi / dim1_stride lo
    g1[7] = 0u;                                         // dim1_stride hi (unused, 2D)

    asm volatile("tensor_load_to_lds %0, %1" :: "s"(g0), "s"(g1) : "memory");
}

// ---------------------------------------------------------------------
// C = A * B^T + bias (NT GEMM, contraction contiguous in both operands).
// A streamed per-wave from global (16 shared rows -> cache hits);
// B tile DMA'd by TDM into double-buffered LDS; split-bf16 WMMA accumulate.
// Block: 256 threads / 8 waves; each wave owns one 16x16 output tile.
// ---------------------------------------------------------------------
__global__ __launch_bounds__(256)
void sae_gemm_nt_tdm(const float* __restrict__ A, int lda,
                     const float* __restrict__ B, int ldb,
                     const float* __restrict__ bias,
                     float* __restrict__ C, int ldc, int K)
{
    __shared__ float sB[2][BROWS * BSTRIDE];            // 2 x 18 KB

    const int wave  = threadIdx.x >> 5;                 // 0..7
    const int lane  = threadIdx.x & 31;                 // wave32
    const int half  = lane >> 4;
    const int l15   = lane & 15;

    const int mBase  = blockIdx.x * 16;                 // token tile (x fastest:
    const int nBlock = blockIdx.y * BROWS;              //  L2-reuses weight slice)
    const int nBase  = nBlock + wave * 16;

    const bool leader = (wave == 0);
    if (leader) {                                       // prologue: 2 tiles in flight
        tdm_load_tile(B + (size_t)nBlock * ldb,      (unsigned)(size_t)&sB[0][0], (unsigned)ldb);
        tdm_load_tile(B + (size_t)nBlock * ldb + KC, (unsigned)(size_t)&sB[1][0], (unsigned)ldb);
    }

    // A: per-lane row m = l15; lanes 0-15 take K 0..7 & 16..23 of each chunk,
    //    lanes 16-31 take K 8..15 & 24..31 (ISA 16-bit A layout).
    const float* aRow = A + (size_t)(mBase + l15) * (size_t)lda;
    const int aOff = half * 8;

    // C/D layout: VGPR v -> (M = v + half*8, N = l15).
    v8f acc;
    const float bv = bias[nBase + l15];
#pragma unroll
    for (int i = 0; i < 8; ++i) acc[i] = bv;

    int buf = 0;
    for (int k0 = 0; k0 < K; k0 += KC) {
        if (leader) {                                   // wait for current tile
            if (k0 + KC < K) __builtin_amdgcn_s_wait_tensorcnt(1);
            else             __builtin_amdgcn_s_wait_tensorcnt(0);
        }
        __syncthreads();                                // publish TDM tile

        // ---- A chunk from global, prefetch one chunk ahead ----
        __builtin_prefetch(aRow + k0 + aOff + KC, 0, 1);
        float af[16];
        {
            const float4* p = reinterpret_cast<const float4*>(aRow + k0 + aOff);
            *reinterpret_cast<float4*>(&af[0]) = p[0];
            *reinterpret_cast<float4*>(&af[4]) = p[1];
            const float4* q = reinterpret_cast<const float4*>(aRow + k0 + aOff + 16);
            *reinterpret_cast<float4*>(&af[8])  = q[0];
            *reinterpret_cast<float4*>(&af[12]) = q[1];
        }

        // ---- B chunk from LDS (bank-skewed rows; ds_load_b128 x4) ----
        // B per-lane: col n = l15; lanes 0-15 take K 0..7 & 16..23, lanes 16-31 K 8..15 & 24..31.
        float bf_[16];
        {
            const float* bRow = &sB[buf][(wave * 16 + l15) * BSTRIDE + half * 8];
            const float4* r = reinterpret_cast<const float4*>(bRow);
            *reinterpret_cast<float4*>(&bf_[0])  = r[0];
            *reinterpret_cast<float4*>(&bf_[4])  = r[1];
            *reinterpret_cast<float4*>(&bf_[8])  = r[4];
            *reinterpret_cast<float4*>(&bf_[12]) = r[5];
        }

        v16bf Ah, Al, Bh, Bl;
#pragma unroll
        for (int e = 0; e < 16; ++e) {
            __bf16 h, l;
            split_bf16(af[e], h, l);  Ah[e] = h; Al[e] = l;
            split_bf16(bf_[e], h, l); Bh[e] = h; Bl[e] = l;
        }

        // acc += Ah*Bh + Ah*Bl + Al*Bh   (error ~ Al*Bl ~ 2^-16 relative)
        acc = __builtin_amdgcn_wmma_f32_16x16x32_bf16(false, Ah, false, Bh,
                                                      (short)0, acc, false, false);
        acc = __builtin_amdgcn_wmma_f32_16x16x32_bf16(false, Ah, false, Bl,
                                                      (short)0, acc, false, false);
        acc = __builtin_amdgcn_wmma_f32_16x16x32_bf16(false, Al, false, Bh,
                                                      (short)0, acc, false, false);

        __syncthreads();                                // all waves done with buf
        if (leader && (k0 + 2 * KC < K))                // refill consumed buffer
            tdm_load_tile(B + (size_t)nBlock * ldb + (k0 + 2 * KC),
                          (unsigned)(size_t)&sB[buf][0], (unsigned)ldb);
        buf ^= 1;
    }

    float* cPtr = C + (size_t)(mBase + half * 8) * (size_t)ldc + (size_t)(nBase + l15);
#pragma unroll
    for (int v = 0; v < 8; ++v)
        cPtr[(size_t)v * (size_t)ldc] = acc[v];
}

// ---------------- exact top-K radix select (in place) ----------------
__device__ __forceinline__ unsigned fkey(float f) {
    unsigned u = __float_as_uint(f);
    return (u & 0x80000000u) ? ~u : (u | 0x80000000u);
}

__global__ __launch_bounds__(256)
void sae_topk_select(float* __restrict__ z)
{
    __shared__ unsigned hist[256];
    __shared__ unsigned s_prefix, s_mask, s_remaining, s_tieTaken;

    float* row = z + (size_t)blockIdx.x * DICT_SIZE;
    const int tid = threadIdx.x;

    if (tid == 0) { s_prefix = 0u; s_mask = 0u; s_remaining = TOPK; }
    __syncthreads();

    for (int pass = 3; pass >= 0; --pass) {             // 4 x 8-bit radix passes
        hist[tid] = 0u;
        __syncthreads();
        const unsigned prefix = s_prefix, mask = s_mask;
        const int shift = pass * 8;
        for (int f = tid; f < DICT_SIZE; f += 256) {
            unsigned key = fkey(row[f]);
            if ((key & mask) == prefix)
                atomicAdd(&hist[(key >> shift) & 255u], 1u);
        }
        __syncthreads();
        if (tid == 0) {
            unsigned rem = s_remaining;
            int bin = 255;
            for (; bin > 0; --bin) {
                if (hist[bin] >= rem) break;
                rem -= hist[bin];
            }
            s_prefix    = prefix | ((unsigned)bin << shift);
            s_mask      = mask   | (0xFFu << shift);
            s_remaining = rem;
        }
        __syncthreads();
    }

    if (tid == 0) s_tieTaken = 0u;
    __syncthreads();
    const unsigned T     = s_prefix;                    // exact threshold key
    const unsigned quota = s_remaining;                 // ties to keep at == T

    for (int f = tid; f < DICT_SIZE; f += 256) {
        const float v = row[f];
        const unsigned key = fkey(v);
        bool keep = key > T;
        if (key == T) keep = (atomicAdd(&s_tieTaken, 1u) < quota);
        if (!keep) row[f] = 0.0f;
    }
}

// ---------------------------------------------------------------------
extern "C" void kernel_launch(void* const* d_in, const int* in_sizes, int n_in,
                              void* d_out, int out_size, void* d_ws, size_t ws_size,
                              hipStream_t stream)
{
    (void)in_sizes; (void)n_in; (void)out_size; (void)d_ws; (void)ws_size;

    const float* x     = (const float*)d_in[0];
    const float* W_enc = (const float*)d_in[1];
    const float* b_enc = (const float*)d_in[2];
    const float* W_dec = (const float*)d_in[3];
    const float* b_dec = (const float*)d_in[4];

    float* x_hat = (float*)d_out;                       // [NTOK, ACT_DIM]
    float* z     = x_hat + (size_t)NTOK * ACT_DIM;      // [NTOK, DICT_SIZE]

    // 1) encode GEMM: z = x * W_enc^T + b_enc  (into sparse_z output region)
    //    grid.x = token tiles (fastest) -> concurrent blocks share the same
    //    2 MB weight slice; x (64 MB) stays L2-resident.
    dim3 encGrid(NTOK / 16, DICT_SIZE / BROWS);         // (256, 256)
    sae_gemm_nt_tdm<<<encGrid, 256, 0, stream>>>(x, ACT_DIM, W_enc, ACT_DIM,
                                                 b_enc, z, DICT_SIZE, ACT_DIM);

    // 2) exact per-row top-64 select, in place
    sae_topk_select<<<NTOK, 256, 0, stream>>>(z);

    // 3) decode GEMM: x_hat = sparse_z * W_dec^T + b_dec
    dim3 decGrid(NTOK / 16, ACT_DIM / BROWS);           // (256, 32)
    sae_gemm_nt_tdm<<<decGrid, 256, 0, stream>>>(z, DICT_SIZE, W_dec, DICT_SIZE,
                                                 b_dec, x_hat, ACT_DIM, DICT_SIZE);
}
